// CrossAttention_15564961480805
// MI455X (gfx1250) — matrix operand order repaired and checked by reading, not compiled, not run on hardware
//
#include <hip/hip_runtime.h>
#include <hip/hip_bf16.h>

typedef __attribute__((ext_vector_type(16))) __bf16 v16bf;
typedef __attribute__((ext_vector_type(8)))  __bf16 v8bf;
typedef __attribute__((ext_vector_type(4)))  __bf16 v4bf;
typedef __attribute__((ext_vector_type(8)))  float  v8f;

#define B_ 16
#define C_ 256
#define NQ 4096      // 64*64
#define NKV 256      // 16*16 after pool
#define HEADS_ 4
#define HD 64
#define EPS_ 1e-5f

static __device__ __forceinline__ v16bf cat16(v8bf lo, v8bf hi) {
  return __builtin_shufflevector(lo, hi, 0,1,2,3,4,5,6,7,8,9,10,11,12,13,14,15);
}

// ---------------------------------------------------------------------------
// 1) q [b,c,4096] fp32 -> qT [b,4096,256] bf16  (tiled LDS transpose)
// ---------------------------------------------------------------------------
__global__ void transpose_q_kernel(const float* __restrict__ q, __bf16* __restrict__ qT) {
  __shared__ float t[32][33];
  int b = blockIdx.z;
  int n0 = blockIdx.x * 32;
  int c0 = blockIdx.y * 32;
  int tx = threadIdx.x, ty = threadIdx.y;
  for (int j = 0; j < 4; ++j) {
    int c = c0 + ty + j * 8;
    t[ty + j * 8][tx] = q[((size_t)b * C_ + c) * NQ + n0 + tx];
  }
  __syncthreads();
  for (int j = 0; j < 4; ++j) {
    int n = n0 + ty + j * 8;
    qT[((size_t)b * NQ + n) * C_ + c0 + tx] = (__bf16)t[tx][ty + j * 8];
  }
}

// ---------------------------------------------------------------------------
// 2) k,v [b,c,32,32] fp32 -> 2x2 mean pool -> transposed bf16 [b,m,c], m=16x16
// ---------------------------------------------------------------------------
__global__ void pool_tr_kernel(const float* __restrict__ k, const float* __restrict__ v,
                               __bf16* __restrict__ kT, __bf16* __restrict__ vT) {
  int c = threadIdx.x;          // 0..255
  int m = blockIdx.x;           // 0..255
  int b = blockIdx.y;
  int h2 = m >> 4, w2 = m & 15;
  size_t base = ((size_t)b * C_ + c) * 1024 + (size_t)(h2 * 2) * 32 + w2 * 2;
  float ks = 0.25f * (k[base] + k[base + 1] + k[base + 32] + k[base + 33]);
  float vs = 0.25f * (v[base] + v[base + 1] + v[base + 32] + v[base + 33]);
  size_t o = ((size_t)b * NKV + m) * C_ + c;
  kT[o] = (__bf16)ks;
  vT[o] = (__bf16)vs;
}

// ---------------------------------------------------------------------------
// 3) GEMM + BN epilogue (shared by q/k/v/final projections).
//    D[n,o] = sum_c A[b,n,c] (* + A2) * W[o,c];  y = D*sc[o] + bi[o]
//    outF : fp32 [b,o,N] ; outT : bf16 [b,n,256] ; outC : bf16 [b,o,N]
// ---------------------------------------------------------------------------
#define GW 4   // waves per block (128 threads)
__global__ void gemm_bn_kernel(const __bf16* __restrict__ A, const __bf16* __restrict__ A2,
                               const float* __restrict__ W, const float* __restrict__ bn,
                               int N, float* __restrict__ outF, __bf16* __restrict__ outT,
                               __bf16* __restrict__ outC) {
  __shared__ __bf16 Wl[64][256];
  int tid = threadIdx.x;
  int b = blockIdx.z;
  int obase = blockIdx.y * 64;
  // stage W tile (fp32 -> bf16) into LDS, float4-vectorized
  for (int l = tid; l < 64 * 256 / 4; l += GW * 32) {
    int r = l >> 6, cc = (l & 63) * 4;
    const float4 f = *(const float4*)(W + (size_t)(obase + r) * 256 + cc);
    v4bf o;
    o[0] = (__bf16)f.x; o[1] = (__bf16)f.y; o[2] = (__bf16)f.z; o[3] = (__bf16)f.w;
    *(v4bf*)(&Wl[r][cc]) = o;
  }
  __syncthreads();

  int wave = tid >> 5, lane = tid & 31;
  int g = lane >> 4, m16 = lane & 15;
  int nbase = (blockIdx.x * GW + wave) * 16;

  v8f acc[4] = {};
  const __bf16* Ab  = A  + ((size_t)b * N + nbase + m16) * 256;
  const __bf16* A2b = A2 ? (A2 + ((size_t)b * N + nbase + m16) * 256) : nullptr;

  for (int kc = 0; kc < 8; ++kc) {
    const v8bf* ap = (const v8bf*)(Ab + kc * 32 + g * 8);
    v8bf lo = ap[0], hi = ap[2];
    if (A2b) {
      const v8bf* a2p = (const v8bf*)(A2b + kc * 32 + g * 8);
      v8bf lo2 = a2p[0], hi2 = a2p[2];
      for (int i = 0; i < 8; ++i) {
        lo[i] = (__bf16)((float)lo[i] + (float)lo2[i]);
        hi[i] = (__bf16)((float)hi[i] + (float)hi2[i]);
      }
    }
    v16bf a = cat16(lo, hi);
    for (int ot = 0; ot < 4; ++ot) {
      const v8bf* wp = (const v8bf*)(&Wl[ot * 16 + m16][kc * 32 + g * 8]);
      v16bf bw = cat16(wp[0], wp[2]);
      acc[ot] = __builtin_amdgcn_wmma_f32_16x16x32_bf16(false, a, false, bw,
                                                        (short)0, acc[ot], false, false);
    }
  }

  for (int ot = 0; ot < 4; ++ot) {
    int o = obase + ot * 16 + m16;
    float sc = bn[o] * rsqrtf(bn[768 + o] + EPS_);
    float bi = bn[256 + o] - bn[512 + o] * sc;
    float y[8];
    for (int r = 0; r < 8; ++r) y[r] = acc[ot][r] * sc + bi;
    int n0 = nbase + 8 * g;
    if (outF) {
      float4 f0 = make_float4(y[0], y[1], y[2], y[3]);
      float4 f1 = make_float4(y[4], y[5], y[6], y[7]);
      float4* p = (float4*)(outF + ((size_t)b * 256 + o) * N + n0);
      p[0] = f0; p[1] = f1;
    }
    if (outT) {
      for (int r = 0; r < 8; ++r)
        outT[((size_t)b * N + n0 + r) * 256 + o] = (__bf16)y[r];
    }
    if (outC) {
      v8bf vb;
      for (int r = 0; r < 8; ++r) vb[r] = (__bf16)y[r];
      *(v8bf*)(outC + ((size_t)b * 256 + o) * N + n0) = vb;
    }
  }
}

// ---------------------------------------------------------------------------
// 4) Attention: one wave per block, 32 q-rows (two 16-row tiles sharing every
//    K/V B-fragment load).  S = scale*Q Kt ; softmax (1 lane per row) ; Y = P V
//    qT [b,4096,256] ; kT [b,256,256] (row=key m) ; vC [b,256(c),256(m)]
//    yT out bf16 [b,4096,256]
// ---------------------------------------------------------------------------
__global__ void attn_kernel(const __bf16* __restrict__ qT, const __bf16* __restrict__ kT,
                            const __bf16* __restrict__ vC, __bf16* __restrict__ yT) {
  __shared__ float sS[32][264];
  int lane = threadIdx.x & 31;
  int g = lane >> 4, m16 = lane & 15;
  int b = blockIdx.z, h = blockIdx.y;
  int nbase = blockIdx.x * 32;

  // preload Q A-fragments: 2 row tiles x 2 k-chunks
  v16bf qa[2][2];
  for (int t = 0; t < 2; ++t) {
    const __bf16* qrow = qT + ((size_t)b * NQ + nbase + t * 16 + m16) * 256 + h * HD;
    for (int kc = 0; kc < 2; ++kc) {
      const v8bf* p = (const v8bf*)(qrow + kc * 32 + g * 8);
      qa[t][kc] = cat16(p[0], p[2]);
    }
  }

  // S = scale * Q K^T  -> LDS scratch (K fragment shared by both row tiles)
  const __bf16* kb = kT + (size_t)b * NKV * 256;
  for (int kt = 0; kt < 16; ++kt) {
    v8f s[2] = {};
    for (int kc = 0; kc < 2; ++kc) {
      const v8bf* kp = (const v8bf*)(kb + (size_t)(kt * 16 + m16) * 256 + h * HD + kc * 32 + g * 8);
      v16bf bk = cat16(kp[0], kp[2]);
      for (int t = 0; t < 2; ++t)
        s[t] = __builtin_amdgcn_wmma_f32_16x16x32_bf16(false, qa[t][kc], false, bk,
                                                       (short)0, s[t], false, false);
    }
    for (int t = 0; t < 2; ++t)
      for (int r = 0; r < 8; ++r)
        sS[t * 16 + r + 8 * g][kt * 16 + m16] = s[t][r] * 0.125f;   // 64^-0.5
  }
  __builtin_amdgcn_wave_barrier();

  // softmax over 256 keys: one lane per row, no cross-lane reduction needed
  {
    float* Sr = &sS[lane][0];
    float mx = -3.4e38f;
    for (int j = 0; j < 256; ++j) mx = fmaxf(mx, Sr[j]);
    float sum = 0.f;
    for (int j = 0; j < 256; ++j) { float e = __expf(Sr[j] - mx); Sr[j] = e; sum += e; }
    float inv = 1.0f / sum;
    for (int j = 0; j < 256; ++j) Sr[j] *= inv;
  }
  __builtin_amdgcn_wave_barrier();

  // Y = P V  (V fragment shared by both row tiles)
  v8f acc[2][4] = {};
  const __bf16* vb = vC + (size_t)b * 256 * NKV;
  for (int mc = 0; mc < 8; ++mc) {
    v16bf pa[2];
    for (int t = 0; t < 2; ++t) {
      const float* pr = &sS[t * 16 + m16][mc * 32 + g * 8];
      for (int i = 0; i < 8; ++i) {
        pa[t][i]     = (__bf16)pr[i];
        pa[t][i + 8] = (__bf16)pr[i + 16];
      }
    }
    for (int dt = 0; dt < 4; ++dt) {
      const v8bf* vp = (const v8bf*)(vb + (size_t)(h * HD + dt * 16 + m16) * NKV + mc * 32 + g * 8);
      v16bf bv = cat16(vp[0], vp[2]);
      for (int t = 0; t < 2; ++t)
        acc[t][dt] = __builtin_amdgcn_wmma_f32_16x16x32_bf16(false, pa[t], false, bv,
                                                             (short)0, acc[t][dt], false, false);
    }
  }
  for (int t = 0; t < 2; ++t)
    for (int dt = 0; dt < 4; ++dt) {
      int c = h * HD + dt * 16 + m16;
      for (int r = 0; r < 8; ++r)
        yT[((size_t)b * NQ + nbase + t * 16 + 8 * g + r) * 256 + c] = (__bf16)acc[t][dt][r];
    }
}

// ---------------------------------------------------------------------------
// 5) depthwise 3x3 'SAME' + BN on q_proj fp32 [b,c,64,64] -> peT bf16 [b,hw,c]
// ---------------------------------------------------------------------------
__global__ void dwconv_pe_kernel(const float* __restrict__ qp, const float* __restrict__ pw,
                                 const float* __restrict__ bn, __bf16* __restrict__ peT) {
  int hw = blockIdx.x * 256 + threadIdx.x;
  int c = blockIdx.y, b = blockIdx.z;
  int y = hw >> 6, x = hw & 63;
  const float* base = qp + ((size_t)b * C_ + c) * NQ;
  const float* w = pw + (size_t)c * 9;
  float s = 0.f;
  for (int dy = -1; dy <= 1; ++dy) {
    int yy = y + dy;
    if (yy < 0 || yy > 63) continue;
    for (int dx = -1; dx <= 1; ++dx) {
      int xx = x + dx;
      if (xx < 0 || xx > 63) continue;
      s += w[(dy + 1) * 3 + (dx + 1)] * base[yy * 64 + xx];
    }
  }
  float sc = bn[c] * rsqrtf(bn[768 + c] + EPS_);
  float bi = bn[256 + c] - bn[512 + c] * sc;
  peT[((size_t)b * NQ + hw) * C_ + c] = (__bf16)(s * sc + bi);
}

// ---------------------------------------------------------------------------
extern "C" void kernel_launch(void* const* d_in, const int* in_sizes, int n_in,
                              void* d_out, int out_size, void* d_ws, size_t ws_size,
                              hipStream_t stream) {
  const float* q       = (const float*)d_in[0];
  const float* k       = (const float*)d_in[1];
  const float* v       = (const float*)d_in[2];
  const float* wq_w    = (const float*)d_in[3];
  const float* wq_bn   = (const float*)d_in[4];
  const float* wk_w    = (const float*)d_in[5];
  const float* wk_bn   = (const float*)d_in[6];
  const float* wv_w    = (const float*)d_in[7];
  const float* wv_bn   = (const float*)d_in[8];
  const float* pe_w    = (const float*)d_in[9];
  const float* pe_bn   = (const float*)d_in[10];
  const float* proj_w  = (const float*)d_in[11];
  const float* proj_bn = (const float*)d_in[12];
  float* out = (float*)d_out;

  char* ws = (char*)d_ws;
  size_t off = 0;
  auto carve = [&](size_t bytes) -> char* {
    char* p = ws + off;
    off = (off + bytes + 255) & ~(size_t)255;
    return p;
  };
  __bf16* qT     = (__bf16*)carve((size_t)B_ * NQ  * C_ * 2);  // [b,n,c]
  float*  q_proj = (float*) carve((size_t)B_ * C_  * NQ * 4);  // [b,c,n] fp32
  __bf16* q_t    = (__bf16*)carve((size_t)B_ * NQ  * C_ * 2);  // [b,n,c]
  __bf16* kTp    = (__bf16*)carve((size_t)B_ * NKV * C_ * 2);  // pooled [b,m,c]
  __bf16* vTp    = (__bf16*)carve((size_t)B_ * NKV * C_ * 2);
  __bf16* k_t    = (__bf16*)carve((size_t)B_ * NKV * C_ * 2);  // [b,m,c] projected
  __bf16* v_c    = (__bf16*)carve((size_t)B_ * C_  * NKV * 2); // [b,c,m] projected
  __bf16* yT     = (__bf16*)carve((size_t)B_ * NQ  * C_ * 2);  // attn out [b,n,c]
  __bf16* peT    = (__bf16*)carve((size_t)B_ * NQ  * C_ * 2);  // pe [b,n,c]
  (void)ws_size; (void)in_sizes; (void)n_in; (void)out_size;

  // 1) transpose q -> bf16 [b,n,c]
  transpose_q_kernel<<<dim3(NQ / 32, C_ / 32, B_), dim3(32, 8), 0, stream>>>(q, qT);
  // 2) pool+transpose k,v
  pool_tr_kernel<<<dim3(NKV, B_), 256, 0, stream>>>(k, v, kTp, vTp);
  // 3) q projection: fp32 [b,c,n] + bf16 [b,n,c]
  gemm_bn_kernel<<<dim3(NQ / 64, 4, B_), GW * 32, 0, stream>>>(
      qT, nullptr, wq_w, wq_bn, NQ, q_proj, q_t, nullptr);
  // 4) k projection: bf16 [b,m,c]
  gemm_bn_kernel<<<dim3(NKV / 64, 4, B_), GW * 32, 0, stream>>>(
      kTp, nullptr, wk_w, wk_bn, NKV, nullptr, k_t, nullptr);
  // 5) v projection: bf16 [b,c,m]
  gemm_bn_kernel<<<dim3(NKV / 64, 4, B_), GW * 32, 0, stream>>>(
      vTp, nullptr, wv_w, wv_bn, NKV, nullptr, nullptr, v_c);
  // 6) attention: 32 q-rows per single-wave block
  attn_kernel<<<dim3(NQ / 32, HEADS_, B_), 32, 0, stream>>>(q_t, k_t, v_c, yT);
  // 7) depthwise 3x3 + BN
  dwconv_pe_kernel<<<dim3(NQ / 256, C_, B_), 256, 0, stream>>>(q_proj, pe_w, pe_bn, peT);
  // 8) final projection on (attn + pe), BN, fp32 output [b,c,64,64]
  gemm_bn_kernel<<<dim3(NQ / 64, 4, B_), GW * 32, 0, stream>>>(
      yT, peT, proj_w, proj_bn, NQ, out, nullptr, nullptr);
}
